// jslab_26319559590178
// MI455X (gfx1250) — matrix operand order repaired and checked by reading, not compiled, not run on hardware
//
#include <hip/hip_runtime.h>

typedef __attribute__((ext_vector_type(16))) _Float16 v16h;
typedef __attribute__((ext_vector_type(8)))  _Float16 v8h;
typedef __attribute__((ext_vector_type(8)))  float    v8f;
typedef __attribute__((ext_vector_type(4)))  unsigned int v4ui;
typedef __attribute__((ext_vector_type(8)))  int      v8si;
typedef __attribute__((ext_vector_type(4)))  int      v4si;

#define HW   (512 * 512)
#define NY   512
#define NX   512
#define TILE 32
#define DT_STEP 1800.0f

// Available only in the device pass; arity differs between toolchains:
// ROCm 7.2 (clang-22): 5 args; amdgpu-toolchain (clang-23+): 6 args.
#if __has_builtin(__builtin_amdgcn_tensor_load_to_lds)
#define USE_TDM 1
#else
#define USE_TDM 0
#endif

__global__ void zero_kernel(float* __restrict__ p, int n) {
  int i = blockIdx.x * blockDim.x + threadIdx.x;
  if (i < n) p[i] = 0.f;
}

#if USE_TDM
// Issue one TDM 2D tile load: 32x32 f32 tile from a 512-stride f32 tensor -> LDS.
// D# packing per CDNA5 ISA 8.3/8.4 (group0: count/lds/global/type, group1: dims/tile/strides).
__device__ __forceinline__ void tdm_load_tile(const float* src, unsigned lds_off) {
  unsigned long long ga = (unsigned long long)(uintptr_t)src;
  v4ui g0;
  g0.x = 1u;                                             // count=1 (valid descriptor)
  g0.y = lds_off;                                        // lds_addr (bytes)
  g0.z = (unsigned)(ga & 0xffffffffu);                   // global_addr[31:0]
  g0.w = (unsigned)((ga >> 32) & 0x01ffffffu) | (2u << 30); // global_addr[56:32] | type=2
  v8si g1;
  g1[0] = (2 << 16);                 // workgroup_mask=0, data_size=2 (4 bytes)
  g1[1] = (NX << 16);                // tensor_dim0[15:0] at bits 63:48
  g1[2] = (int)(((unsigned)NY << 16) | ((unsigned)NX >> 16)); // dim0[31:16] | dim1[15:0]<<16
  g1[3] = (TILE << 16);              // dim1[31:16]=0 | tile_dim0
  g1[4] = TILE;                      // tile_dim1 | tile_dim2(0)<<16
  g1[5] = NX;                        // tensor_dim0_stride[31:0]
  g1[6] = 0;                         // stride0[47:32] | tensor_dim1_stride[15:0]
  g1[7] = 0;                         // tensor_dim1_stride[47:16]
  v4si z4 = {0, 0, 0, 0};
#if __clang_major__ >= 23
  v8si z8 = {0, 0, 0, 0, 0, 0, 0, 0};
  __builtin_amdgcn_tensor_load_to_lds(g0, g1, z4, z4, z8, 0);
#else
  __builtin_amdgcn_tensor_load_to_lds(g0, g1, z4, z4, 0);
#endif
}
#endif

// One fused time substep: conv1(2->16) -> relu -> conv2(16->2) -> relu -> Euler update.
// Block = 256 threads (8 wave32), tile = 32x32 output pixels, halo = 2.
__global__ __launch_bounds__(256)
void slab_substep(const float* __restrict__ Uin, const float* __restrict__ Vin,
                  float* __restrict__ Uout, float* __restrict__ Vout,
                  const float* __restrict__ TAx, const float* __restrict__ TAy,
                  const float* __restrict__ K0, const float* __restrict__ fcp,
                  const float* __restrict__ W1, const float* __restrict__ b1,
                  const float* __restrict__ W2, const float* __restrict__ b2,
                  float* __restrict__ outU, float* __restrict__ outV,
                  int itf, float aa, int row)
{
  __shared__ float    sUV[2 * 1296];                            // 36x36 per channel (2-halo)
  __shared__ __attribute__((aligned(16))) _Float16 sH[1168*16]; // 34x34(+pad) x 16ch
  __shared__ __attribute__((aligned(16))) _Float16 sB1T[16*32]; // conv1 weights [N][K]
  __shared__ __attribute__((aligned(16))) _Float16 sB2T[16*160];// conv2 weights [N][K'], K'=sp*16+ic
  __shared__ float    sb1[16];
  __shared__ float    sb2[2];
  __shared__ float    sDiss[2 * 1024];                          // 32x32 x 2 channels
#if USE_TDM
  __shared__ __attribute__((aligned(16))) float sF[4 * 1024];   // tax0,tax1,tay0,tay1 tiles
#endif

  const int tid  = threadIdx.x;
  const int lane = tid & 31;
  const int wid  = tid >> 5;
  const int hi   = lane >> 4;   // 0: lanes 0-15, 1: lanes 16-31
  const int mloc = lane & 15;

  const int by = blockIdx.y, bx = blockIdx.x;
  const int gy0 = by * TILE - 2;
  const int gx0 = bx * TILE - 2;

#if USE_TDM
  // Wave 0 kicks off async forcing-tile DMA; it overlaps both conv stages.
  if (wid == 0) {
    const float* tx0 = TAx + (size_t)itf * HW + (size_t)(by * TILE) * NX + bx * TILE;
    const float* ty0 = TAy + (size_t)itf * HW + (size_t)(by * TILE) * NX + bx * TILE;
    unsigned fbase = (unsigned)(uintptr_t)&sF[0];   // flat LDS addr low bits = LDS offset
    tdm_load_tile(tx0,      fbase);
    tdm_load_tile(tx0 + HW, fbase + 1 * 4096);
    tdm_load_tile(ty0,      fbase + 2 * 4096);
    tdm_load_tile(ty0 + HW, fbase + 3 * 4096);
  }
#else
  // Warm L2 for the forcing slices consumed at the end of the kernel.
  {
    int i = tid >> 5, j = tid & 31;
    size_t g = (size_t)(by * TILE + i) * NX + (bx * TILE + j);
    const float* tx0 = TAx + (size_t)itf * HW;
    const float* ty0 = TAy + (size_t)itf * HW;
    __builtin_prefetch(tx0 + g, 0, 1);
    __builtin_prefetch(ty0 + g, 0, 1);
    __builtin_prefetch(tx0 + HW + g, 0, 1);
    __builtin_prefetch(ty0 + HW + g, 0, 1);
  }
#endif

  // ---------------- stage A: cooperative LDS fills ----------------
  for (int idx = tid; idx < 1296; idx += 256) {
    int i = idx / 36, j = idx % 36;
    int gi = gy0 + i, gj = gx0 + j;
    bool inb = (gi >= 0) & (gi < NY) & (gj >= 0) & (gj < NX);
    int g = inb ? gi * NX + gj : 0;
    sUV[idx]        = inb ? Uin[g] : 0.f;   // SAME padding -> zeros
    sUV[1296 + idx] = inb ? Vin[g] : 0.f;
  }
  // conv1 weights, [N][K] with K = ic*9 + ky*3 + kx, padded 18->32
  for (int idx = tid; idx < 16 * 32; idx += 256) {
    int n = idx >> 5, k = idx & 31;
    sB1T[idx] = (_Float16)((k < 18) ? W1[n * 18 + k] : 0.f);
  }
  // conv2 weights, [N][K'] with K' = (ky*3+kx)*16 + ic, padded 144->160, N 2->16
  for (int idx = tid; idx < 16 * 160; idx += 256) {
    int n = idx / 160, k = idx % 160;
    int ic = k & 15, sp = k >> 4;
    sB2T[idx] = (_Float16)(((n < 2) & (sp < 9)) ? W2[n * 144 + ic * 9 + sp] : 0.f);
  }
  if (tid < 16) sb1[tid] = b1[tid];
  if (tid < 2)  sb2[tid] = b2[tid];
  __syncthreads();

  const float bias1 = sb1[mloc];
  const float bias2 = sb2[mloc & 1];

  // ---------------- B fragments (weights) in registers ----------------
  // B layout (16-bit, 32x16): element e -> K = (e<8 ? e : e+8) + 8*hi  => two runs of 8 in [N][K].
  v16h bf1;
  {
    const _Float16* p = &sB1T[mloc * 32 + hi * 8];
    v8h q0 = *(const v8h*)p;
    v8h q1 = *(const v8h*)(p + 16);
    bf1 = __builtin_shufflevector(q0, q1, 0,1,2,3,4,5,6,7,8,9,10,11,12,13,14,15);
  }
  v16h bf2[5];
  #pragma unroll
  for (int kt = 0; kt < 5; ++kt) {
    const _Float16* p = &sB2T[mloc * 160 + kt * 32 + hi * 8];
    v8h q0 = *(const v8h*)p;
    v8h q1 = *(const v8h*)(p + 16);
    bf2[kt] = __builtin_shufflevector(q0, q1, 0,1,2,3,4,5,6,7,8,9,10,11,12,13,14,15);
  }

  // ---------------- conv1: h = relu(W1 * [U;V] + b1) on 34x34 region ----------------
  // A layout: element e -> K = (e<8 ? e : e+8) + 8*hi, M = mloc.
  // Offsets for K into flat sUV (ic*1296 + ky*36 + kx) are compile-time.
  auto conv1_group = [&](int g, int base1) {
    const int d1lo[8] = {0, 1, 2, 36, 37, 38, 72, 73};                 // K = 0..7
    const int d1hi[8] = {74, 1296, 1297, 1298, 1332, 1333, 1334, 1368};// K = 8..15
    v16h a;
    #pragma unroll
    for (int e = 0; e < 8; ++e)
      a[e] = (_Float16)sUV[base1 + (hi ? d1hi[e] : d1lo[e])];
    // e=8 -> K=16+8*hi, e=9 -> K=17+8*hi: valid only for hi==0 (K=16,17)
    a[8] = (_Float16)(hi ? 0.f : sUV[base1 + 1369]);
    a[9] = (_Float16)(hi ? 0.f : sUV[base1 + 1370]);
    #pragma unroll
    for (int e = 10; e < 16; ++e) a[e] = (_Float16)0.f;  // K >= 18 always
    v8f c = {0.f, 0.f, 0.f, 0.f, 0.f, 0.f, 0.f, 0.f};
    c = __builtin_amdgcn_wmma_f32_16x16x32_f16(false, a, false, bf1, (short)0, c, false, false);
    // D layout: VGPR r -> M = r + 8*hi, N = mloc; sH padded to 1168 pixels -> branchless.
    #pragma unroll
    for (int r = 0; r < 8; ++r) {
      int M2 = g * 16 + r + hi * 8;
      sH[M2 * 16 + mloc] = (_Float16)fmaxf(c[r] + bias1, 0.f);
    }
  };
  {
    // Incremental pixel->(i,j) tracking: M advances by 16 per group; no div in the loop.
    int M = (wid * 9) * 16 + mloc;            // wave-start pixel, always < 1152 here
    int i0 = M / 34, j0 = M % 34;
    int base1 = i0 * 36 + j0;
    #pragma unroll 1
    for (int iter = 0; iter < 9; ++iter) {
      conv1_group(wid * 9 + iter, base1);
      j0 += 16;
      bool wrap = j0 >= 34;
      base1 += wrap ? 18 : 16;                // +16, +2 extra when row wraps (36-34)
      j0 -= wrap ? 34 : 0;
    }
  }
  if (wid == 0) {                             // remainder group 72 (wave-uniform branch)
    int M = 72 * 16 + mloc;
    int Mc = (M < 1156) ? M : 1155;           // clamp gathers in the padding region
    conv1_group(72, (Mc / 34) * 36 + (Mc % 34));
  }
  __syncthreads();

  // ---------------- conv2: diss = relu(W2 * h + b2) on 32x32 tile ----------------
  // 64 pixel-groups -> exactly 8 per wave; K' = 144 -> 5 K-tiles of 32.
  // A fragment halves are runs of 8 consecutive K' = 8 contiguous channels of one
  // pixel in sH[pixel][ch] => one aligned ds_load_b128 each.
  #pragma unroll 1
  for (int iter = 0; iter < 8; ++iter) {
    int g = wid * 8 + iter;
    int M = g * 16 + mloc;                    // 0..1023
    int i = M >> 5, j = M & 31;
    const _Float16* hb = &sH[(i * 34 + j) * 16 + hi * 8];
    v8f c = {0.f, 0.f, 0.f, 0.f, 0.f, 0.f, 0.f, 0.f};
    #pragma unroll
    for (int kt = 0; kt < 5; ++kt) {
      const int b0 = 2 * kt, b1k = 2 * kt + 1;          // spatial blocks (indep. of hi)
      const int d0 = ((b0 / 3) * 34 + (b0 % 3)) * 16;
      const int d1 = ((b1k / 3) * 34 + (b1k % 3)) * 16;
      v8h r0 = *(const v8h*)(hb + d0);
      v8h r1;
      if (b1k < 9) {
        r1 = *(const v8h*)(hb + d1);
      } else {                                          // kt==4: K' in [144,160) -> zero pad
        r1 = v8h{(_Float16)0.f, (_Float16)0.f, (_Float16)0.f, (_Float16)0.f,
                 (_Float16)0.f, (_Float16)0.f, (_Float16)0.f, (_Float16)0.f};
      }
      v16h a = __builtin_shufflevector(r0, r1, 0,1,2,3,4,5,6,7,8,9,10,11,12,13,14,15);
      c = __builtin_amdgcn_wmma_f32_16x16x32_f16(false, a, false, bf2[kt], (short)0, c, false, false);
    }
    if (mloc < 2) {                           // only N=0,1 are real channels
      #pragma unroll
      for (int r = 0; r < 8; ++r) {
        int M2 = g * 16 + r + hi * 8;
        sDiss[mloc * 1024 + M2] = fmaxf(c[r] + bias2, 0.f);
      }
    }
  }
#if USE_TDM
  if (wid == 0) __builtin_amdgcn_s_wait_tensorcnt(0);   // forcing tiles landed in LDS
#endif
  __syncthreads();

  // ---------------- Euler update + output assembly ----------------
  float Kc  = __expf(K0[0]);
  float fcv = fcp[0];
#if !USE_TDM
  const float* TAx0 = TAx + (size_t)itf * HW;
  const float* TAy0 = TAy + (size_t)itf * HW;
  const float* TAx1 = TAx0 + HW;              // itf+1 always < Nt for this config
  const float* TAy1 = TAy0 + HW;
#endif
  #pragma unroll
  for (int pp = 0; pp < 4; ++pp) {
    int p = tid + pp * 256;
    int i = p >> 5, j = p & 31;
    int gi = by * TILE + i, gj = bx * TILE + j;
    int g = gi * NX + gj;
    float U = sUV[(i + 2) * 36 + (j + 2)];
    float V = sUV[1296 + (i + 2) * 36 + (j + 2)];
#if USE_TDM
    float tax = (1.f - aa) * sF[p]        + aa * sF[1024 + p];
    float tay = (1.f - aa) * sF[2048 + p] + aa * sF[3072 + p];
#else
    float tax = (1.f - aa) * TAx0[g] + aa * TAx1[g];
    float tay = (1.f - aa) * TAy0[g] + aa * TAy1[g];
#endif
    float dU =  fcv * V + Kc * tax - sDiss[p];
    float dV = -fcv * U + Kc * tay - sDiss[1024 + p];
    float U2 = U + DT_STEP * dU;
    float V2 = V + DT_STEP * dV;
    Uout[g] = U2;
    Vout[g] = V2;
    if (row >= 0) {
      outU[(size_t)row * HW + g] = U2;
      outV[(size_t)row * HW + g] = V2;
    }
  }
}

extern "C" void kernel_launch(void* const* d_in, const int* in_sizes, int n_in,
                              void* d_out, int out_size, void* d_ws, size_t ws_size,
                              hipStream_t stream) {
  (void)in_sizes; (void)n_in; (void)ws_size;
  const float* K0  = (const float*)d_in[0];
  const float* fc  = (const float*)d_in[1];
  const float* TAx = (const float*)d_in[2];
  const float* TAy = (const float*)d_in[3];
  const float* W1  = (const float*)d_in[4];
  const float* b1  = (const float*)d_in[5];
  const float* W2  = (const float*)d_in[6];
  const float* b2  = (const float*)d_in[7];
  // d_in[8] = t1 (device int scalar); Nf is derivable from out_size instead.
  const int Nf = out_size / (2 * HW);   // 64

  float* ws   = (float*)d_ws;
  float* bufs[2] = { ws, ws + 2 * HW }; // each buffer: U at +0, V at +HW
  float* outU = (float*)d_out;
  float* outV = outU + (size_t)Nf * HW;

  // U0 = V0 = 0; output row 0 is zeros in the reference's assembly.
  zero_kernel<<<(2 * HW + 255) / 256, 256, 0, stream>>>(bufs[0], 2 * HW);
  zero_kernel<<<(HW + 255) / 256, 256, 0, stream>>>(outU, HW);
  zero_kernel<<<(HW + 255) / 256, 256, 0, stream>>>(outV, HW);

  int cur = 0;
  for (int iout = 0; iout < Nf; ++iout) {
    for (int iin = 0; iin < 2; ++iin) {
      float aa = 0.5f * (float)iin;     // (it % NSUB) / NSUB with NSUB=2
      int row = -1;
      if (iin == 1) {
        // reference assembly: out[1:Nf-1] = S[:Nf-2]; out[Nf-1] = S[Nf-1]; S[Nf-2] dropped
        if (iout <= Nf - 3)      row = iout + 1;
        else if (iout == Nf - 1) row = Nf - 1;
      }
      slab_substep<<<dim3(NX / TILE, NY / TILE), 256, 0, stream>>>(
          bufs[cur], bufs[cur] + HW, bufs[cur ^ 1], bufs[cur ^ 1] + HW,
          TAx, TAy, K0, fc, W1, b1, W2, b2, outU, outV,
          /*itf=*/iout, aa, row);
      cur ^= 1;
    }
  }
}
